// Attention2d_90915867721806
// MI455X (gfx1250) — matrix operand order repaired and checked by reading, not compiled, not run on hardware
//
#include <hip/hip_runtime.h>
#include <hip/hip_bf16.h>

typedef __attribute__((ext_vector_type(16))) __bf16 v16bf;
typedef __attribute__((ext_vector_type(8)))  __bf16 v8bf;
typedef __attribute__((ext_vector_type(8)))  float  v8f;

#define N_IMG   32
#define C_DIM   128
#define S_DIM   1152
#define OUT_DIM 5000
#define NB      64    // neurons per block (4 waves x 16)
#define ST      16    // spatial rows per flash step

// round-to-nearest-even f32 -> bf16 without relying on fp-cast lowering
__device__ __forceinline__ __bf16 bf16_rne(float f) {
    unsigned u = __builtin_bit_cast(unsigned, f);
    u += 0x7FFFu + ((u >> 16) & 1u);
    unsigned short s = (unsigned short)(u >> 16);
    return __builtin_bit_cast(__bf16, s);
}

// build a 16-wide bf16 fragment from two contiguous 8-element (16B) LDS reads
__device__ __forceinline__ v16bf ld_frag(const __bf16* p0, const __bf16* p1) {
    v8bf a = *(const v8bf*)p0;
    v8bf b = *(const v8bf*)p1;
    return __builtin_shufflevector(a, b, 0,1,2,3,4,5,6,7,8,9,10,11,12,13,14,15);
}

__global__ __launch_bounds__(128) void attn2d_fused_flash(
    const float* __restrict__ x,        // (N, C, S)
    const float* __restrict__ features, // (C, OUT)
    const float* __restrict__ nquery,   // (C, OUT)
    const float* __restrict__ bias,     // (OUT)
    const float* __restrict__ pos,      // (C, S)
    float* __restrict__ out)            // (N, OUT)
{
    // LDS staging: Q/F transposed [n_local][c] so B fragments are contiguous runs,
    // x tiles stored [s][c] so A fragments are contiguous runs.
    __shared__ __align__(16) __bf16 qT[NB * C_DIM];   // 16 KB
    __shared__ __align__(16) __bf16 fT[NB * C_DIM];   // 16 KB
    __shared__ __align__(16) __bf16 xeT[ST * C_DIM];  // 4 KB  (x + pos)
    __shared__ __align__(16) __bf16 xfT[ST * C_DIM];  // 4 KB  (x)

    const int img  = blockIdx.x;
    const int nb0  = blockIdx.y * NB;
    const int tid  = threadIdx.x;
    const int lane = tid & 31;          // wave32
    const int wave = tid >> 5;          // 0..3

    // ---- stage Q / F for this block's 64 neurons (bf16, transposed) ----
    for (int j = tid; j < NB * C_DIM; j += 128) {
        int c  = j >> 6;                 // 0..127 (coalesced over n in global)
        int nl = j & 63;
        int n  = nb0 + nl; if (n >= OUT_DIM) n = OUT_DIM - 1;  // clamp (safe dup)
        qT[nl * C_DIM + c] = bf16_rne(nquery  [c * OUT_DIM + n]);
        fT[nl * C_DIM + c] = bf16_rne(features[c * OUT_DIM + n]);
    }
    __syncthreads();

    const int m      = lane & 15;        // A-fragment row (s); D column (neuron)
    const int hiLane = lane >> 4;        // 0: lanes 0-15, 1: lanes 16-31
    const int nlw    = wave * 16 + m;    // local neuron column for this lane
    const int nglob  = nb0 + nlw;
    const long xbase = (long)img * C_DIM * S_DIM;

    // ---- hoist loop-invariant B fragments (Q and F tiles) into registers ----
    // B fragment, this kernel's contiguous k order: lanes<16 hold K 0..15, lanes>=16 hold K 16..31
    v16bf Bq[4], Bf[4];
    #pragma unroll
    for (int k = 0; k < 4; ++k) {
        int bLo = k * 32 + (hiLane ? 16 : 0);
        Bq[k] = ld_frag(&qT[nlw * C_DIM + bLo], &qT[nlw * C_DIM + bLo + 8]);
        Bf[k] = ld_frag(&fT[nlw * C_DIM + bLo], &fT[nlw * C_DIM + bLo + 8]);
    }

    // flash-softmax running state (per lane: one neuron column, 8 of 16 rows)
    float m_run = -__builtin_inff();
    float l_run = 0.f;
    float o_run = 0.f;

    for (int s0 = 0; s0 < S_DIM; s0 += ST) {
        __syncthreads();   // previous step's fragment reads done
        // ---- cooperative stage of x tile: xf (raw) and xe (x + pos) ----
        for (int j = tid; j < C_DIM * ST; j += 128) {
            int c = j >> 4;
            int s = j & 15;              // 16 consecutive threads -> consecutive s
            float xv = x  [xbase + (long)c * S_DIM + s0 + s];
            float pv = pos[(long)c * S_DIM + s0 + s];
            xfT[s * C_DIM + c] = bf16_rne(xv);
            xeT[s * C_DIM + c] = bf16_rne(xv + pv);
        }
        // prefetch next S-tile (global_prefetch_b8; cacheline granular)
        if (s0 + ST < S_DIM) {
            __builtin_prefetch(&x[xbase + (long)tid * S_DIM + s0 + ST], 0, 0);
        }
        __syncthreads();

        // ---- 16x16 score tile and g tile: K = C = 128 -> 4 bf16 WMMAs each ----
        v8f sc = {};   // scores = (x+pos)^T Q
        v8f gg = {};   // g      =  x^T    F
        #pragma unroll
        for (int k = 0; k < 4; ++k) {
            // A fragment, this kernel's contiguous k order (same as B): lanes<16 hold K 0..15, lanes>=16 hold K 16..31
            int aLo = k * 32 + (hiLane ? 16 : 0);
            int aHi = k * 32 + (hiLane ? 24 : 8);

            v16bf A  = ld_frag(&xeT[m * C_DIM + aLo], &xeT[m * C_DIM + aHi]);
            sc = __builtin_amdgcn_wmma_f32_16x16x32_bf16(
                     false, A, false, Bq[k], (short)0, sc, false, false);

            v16bf A2 = ld_frag(&xfT[m * C_DIM + aLo], &xfT[m * C_DIM + aHi]);
            gg = __builtin_amdgcn_wmma_f32_16x16x32_bf16(
                     false, A2, false, Bf[k], (short)0, gg, false, false);
        }

        // ---- flash update: column n is split across lane pair (l, l^16) ----
        float mloc = sc[0];
        #pragma unroll
        for (int r = 1; r < 8; ++r) mloc = fmaxf(mloc, sc[r]);
        float mpair = fmaxf(mloc, __shfl_xor(mloc, 16, 32));
        float m_new = fmaxf(m_run, mpair);
        float alpha = __expf(m_run - m_new);   // 0 on first step (exp(-inf))
        l_run *= alpha;
        o_run *= alpha;
        #pragma unroll
        for (int r = 0; r < 8; ++r) {
            float p = __expf(sc[r] - m_new);
            l_run += p;
            o_run += p * gg[r];
        }
        m_run = m_new;
    }

    // combine the lane pair and write the readout
    float l_tot = l_run + __shfl_xor(l_run, 16, 32);
    float o_tot = o_run + __shfl_xor(o_run, 16, 32);
    if (hiLane == 0 && nglob < OUT_DIM) {
        out[(long)img * OUT_DIM + nglob] = o_tot / l_tot + bias[nglob];
    }
}

extern "C" void kernel_launch(void* const* d_in, const int* in_sizes, int n_in,
                              void* d_out, int out_size, void* d_ws, size_t ws_size,
                              hipStream_t stream) {
    const float* x        = (const float*)d_in[0];
    const float* features = (const float*)d_in[1];
    const float* nquery   = (const float*)d_in[2];
    const float* bias     = (const float*)d_in[3];
    const float* pos      = (const float*)d_in[4];
    float* out = (float*)d_out;

    dim3 grid(N_IMG, (OUT_DIM + NB - 1) / NB);  // (32, 79)
    attn2d_fused_flash<<<grid, 128, 0, stream>>>(x, features, nquery, bias, pos, out);
}
